// Dytrans_45561013076435
// MI455X (gfx1250) — compile-verified, matrix operand order checked by reading, not run on hardware
//
#include <hip/hip_runtime.h>
#include <hip/hip_bf16.h>
#include <cstdint>

// ---------------------------------------------------------------------------
// Shapes (compile-time constants from the reference)
// ---------------------------------------------------------------------------
#define Lq   2048
#define Bq   4
#define Cq   512
#define Hq   8
#define Dq   64
#define Sq   40
#define C4q  2048          // EXP * C
#define Mq   (Bq * Lq)     // 8192 rows in flattened (B,L) space

typedef unsigned short ushort_t;

// bf16 vector types for WMMA
typedef __attribute__((ext_vector_type(16))) __bf16        v16bf;
typedef __attribute__((ext_vector_type(8)))  float         v8f;
typedef __attribute__((ext_vector_type(8)))  unsigned int  v8u;

// ---------------------------------------------------------------------------
// helpers
// ---------------------------------------------------------------------------
__device__ __forceinline__ ushort_t f2bf(float f) {
    unsigned int u = __builtin_bit_cast(unsigned int, f);
    unsigned int r = u + 0x7FFFu + ((u >> 16) & 1u);   // round-to-nearest-even
    return (ushort_t)(r >> 16);
}

__device__ __forceinline__ float dysig(float x, float a, float w, float b) {
    // x * sigmoid(a*x) * w + b
    return x * (1.0f / (1.0f + __expf(-a * x))) * w + b;
}

__device__ __forceinline__ v16bf make_frag(uint4 lo, uint4 hi) {
    v8u u;
    u[0] = lo.x; u[1] = lo.y; u[2] = lo.z; u[3] = lo.w;
    u[4] = hi.x; u[5] = hi.y; u[6] = hi.z; u[7] = hi.w;
    return __builtin_bit_cast(v16bf, u);
}

// async copy of 32 bytes: global -> LDS (covers [0,16) and [16,32) of both)
__device__ __forceinline__ void async_b128x2(unsigned lds_addr, const void* gptr) {
    unsigned long long g = (unsigned long long)(uintptr_t)gptr;
    asm volatile("global_load_async_to_lds_b128 %0, %1, off"
                 :: "v"(lds_addr), "v"(g) : "memory");
    asm volatile("global_load_async_to_lds_b128 %0, %1, off offset:16"
                 :: "v"(lds_addr), "v"(g) : "memory");
}

// ---------------------------------------------------------------------------
// K0: fp32 -> bf16 weight conversion
// ---------------------------------------------------------------------------
__global__ void conv_bf16_kernel(const float* __restrict__ src,
                                 ushort_t* __restrict__ dst, int n) {
    int i = blockIdx.x * 256 + threadIdx.x;
    if (i < n) dst[i] = f2bf(src[i]);
}

// ---------------------------------------------------------------------------
// K1: norm1 + transpose (L,B,C) -> bf16 (B*L, C)
// ---------------------------------------------------------------------------
__global__ void norm1_kernel(const float* __restrict__ query,
                             const float* __restrict__ alpha,
                             const float* __restrict__ nw,
                             const float* __restrict__ nb,
                             ushort_t* __restrict__ xb) {
    size_t i = (size_t)blockIdx.x * 256 + threadIdx.x;  // over L*B*C
    int c   = (int)(i & (Cq - 1));
    size_t r = i >> 9;                                   // l*B + b
    int b = (int)(r & (Bq - 1));
    int l = (int)(r >> 2);
    float a = alpha[0];
    float v = dysig(query[i], a, nw[c], nb[c]);
    xb[((size_t)b * Lq + l) * Cq + c] = f2bf(v);
}

// ---------------------------------------------------------------------------
// K2: WMMA bf16 GEMM  D(M,N) = A(M,K) * W(N,K)^T  (+ epilogues)
//   block = 128 threads (4 waves), block tile 64x64, wave tile 16x64
//   B tile (64 cols x 32 k = 4KB) staged to LDS with async DMA, double-buffered
//   MODE 0: QKV  -> out_f32 in (B,H,L,D) layout (+bias)
//   MODE 1: Wo   -> x = acc + bias + query^T ; y = bf16(norm2(x))
//   MODE 2: FFN1 -> h1 = bf16(relu(acc + bias))
//   MODE 3: FFN2 -> d_out[(l*B+b)*C+c] = acc + bias + x
// ---------------------------------------------------------------------------
template <int MODE>
__global__ __launch_bounds__(128)
void gemm_bf16_kernel(const ushort_t* __restrict__ A,
                      const ushort_t* __restrict__ W,
                      const float* __restrict__ bias,
                      int K, int N,
                      float* __restrict__ out_f32,
                      ushort_t* __restrict__ out_bf16,
                      const float* __restrict__ resid,
                      const float* __restrict__ alpha,
                      const float* __restrict__ nw,
                      const float* __restrict__ nb) {
    __shared__ ushort_t bsm[2][64 * 32];     // double-buffered B tile

    const int tid  = threadIdx.x;
    const int w    = tid >> 5;               // wave in block: 0..3
    const int lane = tid & 31;
    const int r    = lane & 15;
    const int half = lane >> 4;              // 0: K 0-7/16-23, 1: K 8-15/24-31

    const int rowBase = blockIdx.x * 64 + w * 16;
    const int colBase = blockIdx.y * 64;

    v8f acc[4];
#pragma unroll
    for (int t = 0; t < 4; ++t) acc[t] = (v8f){0,0,0,0,0,0,0,0};

    const ushort_t* aRow = A + (size_t)(rowBase + r) * K + half * 8;

    // this thread's 32-byte slice of the B tile: row tid/2, 32B segment tid&1
    const ushort_t* bsrc = W + (size_t)(colBase + (tid >> 1)) * K + (tid & 1) * 16;
    const unsigned ldsSlice0 = (unsigned)(uintptr_t)(void*)&bsm[0][0] + (unsigned)tid * 32u;
    const unsigned ldsSlice1 = (unsigned)(uintptr_t)(void*)&bsm[1][0] + (unsigned)tid * 32u;

    // prologue: stage k0=0 into buffer 0
    async_b128x2(ldsSlice0, bsrc);

    int cur = 0;
    for (int k0 = 0; k0 < K; k0 += 32) {
        const bool more = (k0 + 32) < K;     // uniform across block
        if (more) {
            async_b128x2(cur ? ldsSlice0 : ldsSlice1, bsrc + (k0 + 32));
            asm volatile("s_wait_asynccnt 0x2" ::: "memory");  // older pair done
        } else {
            asm volatile("s_wait_asynccnt 0x0" ::: "memory");
        }
        __syncthreads();                      // publish LDS writes block-wide

        const ushort_t* pa = aRow + k0;
        uint4 alo = *(const uint4*)(pa);
        uint4 ahi = *(const uint4*)(pa + 16);
        __builtin_prefetch((const void*)(pa + 32), 0, 1);   // global_prefetch
        v16bf afrag = make_frag(alo, ahi);

        const ushort_t* bbase = cur ? &bsm[1][0] : &bsm[0][0];
#pragma unroll
        for (int t = 0; t < 4; ++t) {
            const ushort_t* pb = bbase + (16 * t + r) * 32 + half * 8;
            uint4 blo = *(const uint4*)(pb);       // ds_load_b128
            uint4 bhi = *(const uint4*)(pb + 16);
            v16bf bfrag = make_frag(blo, bhi);
            acc[t] = __builtin_amdgcn_wmma_f32_16x16x32_bf16(
                         false, afrag, false, bfrag, (short)0, acc[t], false, false);
        }
        __syncthreads();                      // protect buffer overwritten next iter
        cur ^= 1;
    }

    // epilogue: element j -> row = rowBase + 8*half + j, col = colBase + 16*t + r
    float al = 0.f;
    if (MODE == 1) al = alpha[0];
#pragma unroll
    for (int t = 0; t < 4; ++t) {
        const int col = colBase + 16 * t + r;
        const float bs = bias[col];
#pragma unroll
        for (int j = 0; j < 8; ++j) {
            const int row = rowBase + 8 * half + j;
            float v = acc[t][j] + bs;
            if (MODE == 0) {
                int b = row >> 11, l = row & (Lq - 1);
                int h = col >> 6,  d = col & (Dq - 1);
                out_f32[(((size_t)b * Hq + h) * Lq + l) * Dq + d] = v;
            } else if (MODE == 1) {
                int b = row >> 11, l = row & (Lq - 1);
                float x = v + resid[((size_t)l * Bq + b) * Cq + col];
                out_f32[(size_t)row * Cq + col] = x;
                out_bf16[(size_t)row * Cq + col] = f2bf(dysig(x, al, nw[col], nb[col]));
            } else if (MODE == 2) {
                out_bf16[(size_t)row * N + col] = f2bf(fmaxf(v, 0.0f));
            } else {  // MODE 3
                int b = row >> 11, l = row & (Lq - 1);
                float o = v + resid[(size_t)row * Cq + col];
                out_f32[((size_t)l * Bq + b) * Cq + col] = o;
            }
        }
    }
}

// ---------------------------------------------------------------------------
// K3: sampled-score metric M = max_s(q.k_s) - mean_s(q.k_s)   per (b,h,l)
// ---------------------------------------------------------------------------
__global__ __launch_bounds__(64)
void score_kernel(const float* __restrict__ q, const float* __restrict__ k,
                  const int* __restrict__ sidx, float* __restrict__ Msc) {
    int gid = blockIdx.x;                 // bh*L + l
    int l  = gid & (Lq - 1);
    int bh = gid >> 11;
    __shared__ float qs[Dq];
    __shared__ float sv[64];
    int t = threadIdx.x;
    qs[t] = q[((size_t)bh * Lq + l) * Dq + t];
    __syncthreads();
    float dot = 0.f;
    if (t < Sq) {
        int idx = sidx[l * Sq + t];
        const float* kr = k + ((size_t)bh * Lq + idx) * Dq;
#pragma unroll
        for (int d = 0; d < Dq; ++d) dot += qs[d] * kr[d];
    }
    sv[t] = dot;
    __syncthreads();
    if (t == 0) {
        float mx = -3.4e38f, sm = 0.f;
        for (int s = 0; s < Sq; ++s) { float v = sv[s]; mx = fmaxf(mx, v); sm += v; }
        Msc[gid] = mx - sm * (1.0f / Sq);
    }
}

// ---------------------------------------------------------------------------
// K4: iterative top-40 over L per (b,h)
// ---------------------------------------------------------------------------
__global__ __launch_bounds__(256)
void topk_kernel(const float* __restrict__ Msc, int* __restrict__ mtop) {
    int bh = blockIdx.x;
    __shared__ float sval[Lq];
    __shared__ float rmax[256];
    __shared__ int   ridx[256];
    int t = threadIdx.x;
    for (int i = t; i < Lq; i += 256) sval[i] = Msc[(size_t)bh * Lq + i];
    __syncthreads();
    for (int it = 0; it < Sq; ++it) {
        float bm = -3.4e38f; int bi = 0;
        for (int i = t; i < Lq; i += 256) {
            float v = sval[i];
            if (v > bm) { bm = v; bi = i; }
        }
        rmax[t] = bm; ridx[t] = bi;
        __syncthreads();
        for (int s = 128; s > 0; s >>= 1) {
            if (t < s && rmax[t + s] > rmax[t]) { rmax[t] = rmax[t + s]; ridx[t] = ridx[t + s]; }
            __syncthreads();
        }
        if (t == 0) { mtop[bh * Sq + it] = ridx[0]; sval[ridx[0]] = -3.4e38f; }
        __syncthreads();
    }
}

// ---------------------------------------------------------------------------
// K5: v mean over L per (b,h,d)
// ---------------------------------------------------------------------------
__global__ __launch_bounds__(64)
void vmean_kernel(const float* __restrict__ v, float* __restrict__ vm) {
    int bh = blockIdx.x, d = threadIdx.x;
    float s = 0.f;
    for (int l = 0; l < Lq; ++l) s += v[((size_t)bh * Lq + l) * Dq + d];
    vm[bh * Dq + d] = s * (1.0f / Lq);
}

// ---------------------------------------------------------------------------
// K6: softmax(q_red . k / sqrt(D)) @ v  per selected query
// ---------------------------------------------------------------------------
__global__ __launch_bounds__(256)
void attn_kernel(const float* __restrict__ q, const float* __restrict__ k,
                 const float* __restrict__ v, const int* __restrict__ mtop,
                 float* __restrict__ upd) {
    int blk = blockIdx.x;                  // bh*S + u
    int u = blk % Sq, bh = blk / Sq;
    int qi = mtop[bh * Sq + u];
    __shared__ float qs[Dq];
    __shared__ float red[256];
    __shared__ float us[Dq];
    int t = threadIdx.x;
    if (t < Dq) { qs[t] = q[((size_t)bh * Lq + qi) * Dq + t]; us[t] = 0.f; }
    __syncthreads();

    float sc[8];
    float lmax = -3.4e38f;
#pragma unroll
    for (int i = 0; i < 8; ++i) {
        int l = t + i * 256;
        const float* kr = k + ((size_t)bh * Lq + l) * Dq;
        float dot = 0.f;
#pragma unroll
        for (int d = 0; d < Dq; ++d) dot += qs[d] * kr[d];
        dot *= 0.125f;                    // 1/sqrt(64)
        sc[i] = dot;
        lmax = fmaxf(lmax, dot);
    }
    red[t] = lmax; __syncthreads();
    for (int s = 128; s > 0; s >>= 1) { if (t < s) red[t] = fmaxf(red[t], red[t + s]); __syncthreads(); }
    float gmax = red[0]; __syncthreads();

    float lsum = 0.f;
#pragma unroll
    for (int i = 0; i < 8; ++i) { float e = __expf(sc[i] - gmax); sc[i] = e; lsum += e; }
    red[t] = lsum; __syncthreads();
    for (int s = 128; s > 0; s >>= 1) { if (t < s) red[t] += red[t + s]; __syncthreads(); }
    float inv = 1.0f / red[0];

    float acc[Dq];
#pragma unroll
    for (int d = 0; d < Dq; ++d) acc[d] = 0.f;
#pragma unroll
    for (int i = 0; i < 8; ++i) {
        int l = t + i * 256;
        float p = sc[i] * inv;
        const float* vr = v + ((size_t)bh * Lq + l) * Dq;
#pragma unroll
        for (int d = 0; d < Dq; ++d) acc[d] += p * vr[d];
    }
#pragma unroll
    for (int d = 0; d < Dq; ++d) atomicAdd(&us[d], acc[d]);
    __syncthreads();
    if (t < Dq) upd[((size_t)bh * Sq + u) * Dq + t] = us[t];
}

// ---------------------------------------------------------------------------
// K7: ctx = broadcast(v_mean) as bf16 (B*L, C); then scatter upd rows
// ---------------------------------------------------------------------------
__global__ void ctxfill_kernel(const float* __restrict__ vm, ushort_t* __restrict__ ctx) {
    size_t i = (size_t)blockIdx.x * 256 + threadIdx.x;  // over B*L*C
    int c = (int)(i & (Cq - 1));
    size_t n = i >> 9;
    int b = (int)(n >> 11);
    int h = c >> 6, d = c & (Dq - 1);
    ctx[i] = f2bf(vm[(b * Hq + h) * Dq + d]);
}

__global__ __launch_bounds__(64)
void scatter_kernel(const float* __restrict__ upd, const int* __restrict__ mtop,
                    ushort_t* __restrict__ ctx) {
    int blk = blockIdx.x;                  // bh*S + u
    int u = blk % Sq, bh = blk / Sq;
    int b = bh >> 3, h = bh & 7;
    int l = mtop[bh * Sq + u];
    int d = threadIdx.x;
    ctx[((size_t)b * Lq + l) * Cq + h * Dq + d] = f2bf(upd[((size_t)bh * Sq + u) * Dq + d]);
}

// ---------------------------------------------------------------------------
// launch
// ---------------------------------------------------------------------------
extern "C" void kernel_launch(void* const* d_in, const int* in_sizes, int n_in,
                              void* d_out, int out_size, void* d_ws, size_t ws_size,
                              hipStream_t stream) {
    const float* query = (const float*)d_in[0];
    const int*   sidx  = (const int*)  d_in[1];
    const float* Wqf   = (const float*)d_in[2];
    const float* bqf   = (const float*)d_in[3];
    const float* Wkf   = (const float*)d_in[4];
    const float* bkf   = (const float*)d_in[5];
    const float* Wvf   = (const float*)d_in[6];
    const float* bvf   = (const float*)d_in[7];
    const float* Wof   = (const float*)d_in[8];
    const float* bof   = (const float*)d_in[9];
    const float* n1a   = (const float*)d_in[10];
    const float* n1w   = (const float*)d_in[11];
    const float* n1b   = (const float*)d_in[12];
    const float* n2a   = (const float*)d_in[13];
    const float* n2w   = (const float*)d_in[14];
    const float* n2b   = (const float*)d_in[15];
    const float* W1f   = (const float*)d_in[16];
    const float* b1f   = (const float*)d_in[17];
    const float* W2f   = (const float*)d_in[18];
    const float* b2f   = (const float*)d_in[19];
    float* outp        = (float*)d_out;

    // workspace carving (256B aligned)
    char* base = (char*)d_ws;
    size_t off = 0;
    auto carve = [&](size_t bytes) -> char* {
        char* r = base + off;
        off += (bytes + 255) & ~(size_t)255;
        return r;
    };
    ushort_t* xb   = (ushort_t*)carve((size_t)Mq * Cq * 2);       // bf16 activations; later reused as ctx
    ushort_t* wqb  = (ushort_t*)carve((size_t)Cq * Cq * 2);
    ushort_t* wkb  = (ushort_t*)carve((size_t)Cq * Cq * 2);
    ushort_t* wvb  = (ushort_t*)carve((size_t)Cq * Cq * 2);
    ushort_t* wob  = (ushort_t*)carve((size_t)Cq * Cq * 2);
    ushort_t* w1b  = (ushort_t*)carve((size_t)C4q * Cq * 2);
    ushort_t* w2b  = (ushort_t*)carve((size_t)Cq * C4q * 2);
    float* qbuf    = (float*)carve((size_t)Bq * Hq * Lq * Dq * 4); // later reused as h1 (spans q+k)
    float* kbuf    = (float*)carve((size_t)Bq * Hq * Lq * Dq * 4);
    float* vbuf    = (float*)carve((size_t)Bq * Hq * Lq * Dq * 4);
    float* Msc     = (float*)carve((size_t)Bq * Hq * Lq * 4);
    int*   mtop    = (int*)  carve((size_t)Bq * Hq * Sq * 4);
    float* vm      = (float*)carve((size_t)Bq * Hq * Dq * 4);
    float* updb    = (float*)carve((size_t)Bq * Hq * Sq * Dq * 4);
    float* xbuf    = (float*)carve((size_t)Mq * Cq * 4);
    ushort_t* ybuf = (ushort_t*)carve((size_t)Mq * Cq * 2);
    (void)kbuf;
    ushort_t* ctx = xb;               // alias: xb dead after QKV GEMMs
    ushort_t* h1  = (ushort_t*)qbuf;  // alias: q/k dead after attention (32 MB)

    // K0: weight conversions
    conv_bf16_kernel<<<(Cq * Cq + 255) / 256, 256, 0, stream>>>(Wqf, wqb, Cq * Cq);
    conv_bf16_kernel<<<(Cq * Cq + 255) / 256, 256, 0, stream>>>(Wkf, wkb, Cq * Cq);
    conv_bf16_kernel<<<(Cq * Cq + 255) / 256, 256, 0, stream>>>(Wvf, wvb, Cq * Cq);
    conv_bf16_kernel<<<(Cq * Cq + 255) / 256, 256, 0, stream>>>(Wof, wob, Cq * Cq);
    conv_bf16_kernel<<<(C4q * Cq + 255) / 256, 256, 0, stream>>>(W1f, w1b, C4q * Cq);
    conv_bf16_kernel<<<(Cq * C4q + 255) / 256, 256, 0, stream>>>(W2f, w2b, Cq * C4q);

    // K1: norm1 + transpose to bf16
    norm1_kernel<<<(Mq * Cq) / 256, 256, 0, stream>>>(query, n1a, n1w, n1b, xb);

    // K2 (MODE 0): q/k/v projections via WMMA
    dim3 gQKV(Mq / 64, Cq / 64);
    gemm_bf16_kernel<0><<<gQKV, 128, 0, stream>>>(xb, wqb, bqf, Cq, Cq, qbuf, nullptr, nullptr, nullptr, nullptr, nullptr);
    gemm_bf16_kernel<0><<<gQKV, 128, 0, stream>>>(xb, wkb, bkf, Cq, Cq, kbuf, nullptr, nullptr, nullptr, nullptr, nullptr);
    gemm_bf16_kernel<0><<<gQKV, 128, 0, stream>>>(xb, wvb, bvf, Cq, Cq, vbuf, nullptr, nullptr, nullptr, nullptr, nullptr);

    // K3..K6: ProbSparse attention core (fp32 VALU)
    score_kernel<<<Bq * Hq * Lq, 64, 0, stream>>>(qbuf, kbuf, sidx, Msc);
    topk_kernel<<<Bq * Hq, 256, 0, stream>>>(Msc, mtop);
    vmean_kernel<<<Bq * Hq, 64, 0, stream>>>(vbuf, vm);
    attn_kernel<<<Bq * Hq * Sq, 256, 0, stream>>>(qbuf, kbuf, vbuf, mtop, updb);

    // K7: build ctx (bf16) = broadcast v-mean, then scatter updated rows
    ctxfill_kernel<<<(Mq * Cq) / 256, 256, 0, stream>>>(vm, ctx);
    scatter_kernel<<<Bq * Hq * Sq, 64, 0, stream>>>(updb, mtop, ctx);

    // K2 (MODE 1): output projection + residual + norm2 -> x (f32), y (bf16)
    gemm_bf16_kernel<1><<<gQKV, 128, 0, stream>>>(ctx, wob, bof, Cq, Cq, xbuf, ybuf, query, n2a, n2w, n2b);

    // K2 (MODE 2): FFN up + ReLU -> h1 (bf16)
    dim3 gF1(Mq / 64, C4q / 64);
    gemm_bf16_kernel<2><<<gF1, 128, 0, stream>>>(ybuf, w1b, b1f, Cq, C4q, nullptr, h1, nullptr, nullptr, nullptr, nullptr);

    // K2 (MODE 3): FFN down + residual + transpose to (L,B,C)
    dim3 gF2(Mq / 64, Cq / 64);
    gemm_bf16_kernel<3><<<gF2, 128, 0, stream>>>(h1, w2b, b2f, C4q, Cq, outp, nullptr, xbuf, nullptr, nullptr, nullptr);
}
